// DeConv2d_62964220559741
// MI455X (gfx1250) — matrix operand order repaired and checked by reading, not compiled, not run on hardware
//
#include <hip/hip_runtime.h>
#include <hip/hip_bf16.h>

// CDNA5 / gfx1250 fused "DeConv2d-as-per-channel-MLP" using bf16 WMMA.
// Layers 1/2 on v_wmma_f32_16x16x32_bf16 (fp32 accumulate), layer 3 fused
// into the L2 epilogue in registers. ~52 GFLOP total, compute-bound.

typedef __bf16 v16bf __attribute__((ext_vector_type(16)));
typedef float  v8f   __attribute__((ext_vector_type(8)));

#define IC_    128
#define H_     256
#define OC_    32
#define P_TOT  8192
#define MTILE  64
#define LDH    264   // h1 LDS row pitch in bf16 elements (256 + 8 pad -> 528B rows)

union Frag16 {
    v16bf b;
    uint4 q[2];
};

// fp32 -> bf16, round-half-away: one VALU add, then the (ushort)(u>>16) store
// pattern-matches to ds_store_b16_d16_hi (zero extra cost for the extract).
__device__ __forceinline__ unsigned short bf16_rna(float f) {
    return (unsigned short)((__float_as_uint(f) + 0x8000u) >> 16);
}

// ---------------------------------------------------------------------------
// Generic batched transpose + fp32->bf16 convert: src [B][R][C] f32 row-major
// -> dst [B][C][R] bf16 row-major. Block (32,8), 32x32 tiles via LDS.
// ---------------------------------------------------------------------------
__global__ void transcvt_kernel(const float* __restrict__ src,
                                unsigned short* __restrict__ dst,
                                int R, int C) {
    __shared__ unsigned short tile[32][33];
    const size_t bo = (size_t)blockIdx.z * (size_t)R * (size_t)C;
    src += bo;
    dst += bo;
    const int c0 = blockIdx.x * 32, r0 = blockIdx.y * 32;
    const int tx = threadIdx.x, ty = threadIdx.y;
#pragma unroll
    for (int i = 0; i < 4; ++i) {
        int r = r0 + ty + i * 8;
        tile[ty + i * 8][tx] = bf16_rna(src[(size_t)r * C + c0 + tx]);
    }
    __syncthreads();
#pragma unroll
    for (int i = 0; i < 4; ++i) {
        int c = c0 + ty + i * 8;
        dst[(size_t)c * R + r0 + tx] = tile[tx][ty + i * 8];
    }
}

// ---------------------------------------------------------------------------
// Fused per-channel 3-layer MLP. One workgroup = (64 pixels, 1 out-channel).
// 8 waves: wave w -> M-tile mt = w>>1 (16 rows), N-group ng = w&1 (128 cols).
// ---------------------------------------------------------------------------
__global__ __launch_bounds__(256, 2)
void fused_mlp_kernel(const unsigned short* __restrict__ Ag,   // [8192][128] bf16 (pixel-major x)
                      const unsigned short* __restrict__ W1t,  // [32][256][128] bf16 (n-major)
                      const unsigned short* __restrict__ W2t,  // [32][256][256] bf16 (n-major)
                      const float* __restrict__ b1,            // [32][256]
                      const float* __restrict__ b2,            // [32][256]
                      const float* __restrict__ W3,            // [32][256][4]
                      const float* __restrict__ b3,            // [32][4]
                      float* __restrict__ out)                 // [8][32][64][64]
{
    __shared__ unsigned short h1[MTILE * LDH];
    __shared__ float obuf[MTILE * 4];

    const int o    = blockIdx.y;
    const int p0   = blockIdx.x * MTILE;
    const int tid  = threadIdx.x;
    const int lane = tid & 31;
    const int wv   = tid >> 5;
    const int mt   = wv >> 1;           // 0..3
    const int ng   = wv & 1;            // 0..1
    const int m0   = mt * 16;
    const int ln   = lane & 15;         // n (B/C cols) or m (A rows)
    const int half = lane >> 4;

    obuf[tid] = 0.0f;

    const unsigned short* w1o = W1t + (size_t)o * H_ * IC_;
    const unsigned short* w2o = W2t + (size_t)o * H_ * H_;
    const float* b1o = b1 + o * H_;
    const float* b2o = b2 + o * H_;
    const float* w3o = W3 + o * H_ * 4;

    // ------------------ Layer 1: [64x128] x [128x256] -> h1 (LDS, bf16) ----
    Frag16 aF[4];
    {
        const unsigned short* abase = Ag + (size_t)(p0 + m0 + ln) * IC_ + half * 8;
#pragma unroll
        for (int k = 0; k < 4; ++k) {
            const unsigned short* p = abase + k * 32;
            aF[k].q[0] = *(const uint4*)p;
            aF[k].q[1] = *(const uint4*)(p + 16);
        }
    }
    for (int j = 0; j < 8; ++j) {
        const int n0 = ng * 128 + j * 16;
        v8f acc = {};
        const unsigned short* bbase = w1o + (size_t)(n0 + ln) * IC_ + half * 8;
#pragma unroll
        for (int k = 0; k < 4; ++k) {
            Frag16 bF;
            const unsigned short* p = bbase + k * 32;
            bF.q[0] = *(const uint4*)p;
            bF.q[1] = *(const uint4*)(p + 16);
            acc = __builtin_amdgcn_wmma_f32_16x16x32_bf16(
                false, aF[k].b, false, bF.b, (short)0, acc, false, false);
        }
        const float bv = b1o[n0 + ln];
        unsigned short* hrow = &h1[(m0 + half * 8) * LDH + n0 + ln];
#pragma unroll
        for (int r = 0; r < 8; ++r) {
            const float v = fmaxf(acc[r] + bv, 0.0f);        // ReLU
            hrow[r * LDH] = bf16_rna(v);                     // 1 add + d16_hi store
        }
    }
    __syncthreads();

    // --------- Layer 2 (WMMA) + Layer 3 contraction fused in registers -----
    float pout[8][4];
#pragma unroll
    for (int r = 0; r < 8; ++r)
#pragma unroll
        for (int kk = 0; kk < 4; ++kk) pout[r][kk] = 0.0f;

    const unsigned short* h1base = h1 + (m0 + ln) * LDH + half * 8;
    for (int j = 0; j < 8; ++j) {
        const int n0 = ng * 128 + j * 16;
        v8f acc = {};
        const unsigned short* bbase = w2o + (size_t)(n0 + ln) * H_ + half * 8;
#pragma unroll
        for (int k = 0; k < 8; ++k) {
            Frag16 aFl, bF;
            const unsigned short* pa = h1base + k * 32;
            aFl.q[0] = *(const uint4*)pa;
            aFl.q[1] = *(const uint4*)(pa + 16);
            const unsigned short* pb = bbase + k * 32;
            bF.q[0] = *(const uint4*)pb;
            bF.q[1] = *(const uint4*)(pb + 16);
            acc = __builtin_amdgcn_wmma_f32_16x16x32_bf16(
                false, aFl.b, false, bF.b, (short)0, acc, false, false);
        }
        const float  bv  = b2o[n0 + ln];
        const float4 w3v = *(const float4*)(w3o + (n0 + ln) * 4);
#pragma unroll
        for (int r = 0; r < 8; ++r) {
            const float v = fmaxf(acc[r] + bv, 0.0f);        // ReLU -> h2 element
            pout[r][0] += v * w3v.x;
            pout[r][1] += v * w3v.y;
            pout[r][2] += v * w3v.z;
            pout[r][3] += v * w3v.w;
        }
    }

    // Reduce over the 16-lane n-dimension (stays within each half-wave).
#pragma unroll
    for (int mask = 1; mask < 16; mask <<= 1) {
#pragma unroll
        for (int r = 0; r < 8; ++r)
#pragma unroll
            for (int kk = 0; kk < 4; ++kk)
                pout[r][kk] += __shfl_xor(pout[r][kk], mask, 32);
    }
    // Combine the two N-group waves per M-tile via small LDS atomics.
    if (ln < 4) {
#pragma unroll
        for (int r = 0; r < 8; ++r)
            atomicAdd(&obuf[(m0 + half * 8 + r) * 4 + ln], pout[r][ln]);
    }
    __syncthreads();

    // ------------------ epilogue: +b3, 2x2 pixel-shuffle scatter -----------
    {
        const int pl = tid >> 2;            // local pixel 0..63
        const int kk = tid & 3;
        const int p  = p0 + pl;
        const int n  = p >> 10;
        const int hw = p & 1023;
        const int h  = hw >> 5;
        const int w  = hw & 31;
        const int kh = kk >> 1;
        const int kw = kk & 1;
        const float val = obuf[pl * 4 + kk] + b3[o * 4 + kk];
        out[((size_t)(n * OC_ + o) * 64 + (h * 2 + kh)) * 64 + (w * 2 + kw)] = val;
    }
}

// ---------------------------------------------------------------------------
extern "C" void kernel_launch(void* const* d_in, const int* in_sizes, int n_in,
                              void* d_out, int out_size, void* d_ws, size_t ws_size,
                              hipStream_t stream) {
    (void)in_sizes; (void)n_in; (void)out_size; (void)ws_size;
    const float* x  = (const float*)d_in[0];
    const float* W1 = (const float*)d_in[1];
    const float* b1 = (const float*)d_in[2];
    const float* W2 = (const float*)d_in[3];
    const float* b2 = (const float*)d_in[4];
    const float* W3 = (const float*)d_in[5];
    const float* b3 = (const float*)d_in[6];
    float* out = (float*)d_out;

    // workspace layout (bf16 operands, 8 MiB total)
    unsigned short* Ag  = (unsigned short*)d_ws;                          // [8192][128]   2 MiB
    unsigned short* W1t = (unsigned short*)((char*)d_ws + (2u << 20));    // [32][256][128] 2 MiB
    unsigned short* W2t = (unsigned short*)((char*)d_ws + (4u << 20));    // [32][256][256] 4 MiB

    dim3 tb(32, 8);
    // x [8][128][1024] -> A [8][1024][128]  (pixel-major, bf16)
    transcvt_kernel<<<dim3(1024 / 32, 128 / 32, 8),  tb, 0, stream>>>(x,  Ag,  128, 1024);
    // W1 [32][128][256] -> W1t [32][256][128]
    transcvt_kernel<<<dim3(256 / 32,  128 / 32, 32), tb, 0, stream>>>(W1, W1t, 128, 256);
    // W2 [32][256][256] -> W2t [32][256][256] (transposed)
    transcvt_kernel<<<dim3(256 / 32,  256 / 32, 32), tb, 0, stream>>>(W2, W2t, 256, 256);

    fused_mlp_kernel<<<dim3(P_TOT / MTILE, OC_), 256, 0, stream>>>(
        Ag, W1t, W2t, b1, b2, W3, b3, out);
}
